// MambaLayer_32744830664823
// MI455X (gfx1250) — compile-verified, matrix operand order checked by reading, not run on hardware
//
#include <hip/hip_runtime.h>
#include <cstdint>
#include <cmath>

// ---------------- problem constants ----------------
#define B_      4
#define C_IN    64
#define H_      128
#define W_      128
#define HW      16384
#define DIM     16
#define D_STATE 16
#define D_MODEL 128
#define D_INNER 256
#define DT_RANK 8
#define M_TOK   65536          // B_*HW tokens
#define EPSC    1e-5f

// chunked scan
#define CHUNK   128
#define NCHUNK  (HW / CHUNK)   // 128

// pooled scales 33/65/97 (scale 1 handled as global mean)
#define S1 33
#define S2 65
#define S3 97
#define A1 (S1*S1)             // 1089
#define A2 (S2*S2)             // 4225
#define A3 (S3*S3)             // 9409
#define APC (A1+A2+A3)         // 14723

typedef __attribute__((ext_vector_type(16))) _Float16 v16h;
typedef __attribute__((ext_vector_type(8)))  float    v8f;

// ---------------- fp32 -> fp16 convert ----------------
__global__ void f32_to_f16_kernel(const float* __restrict__ in,
                                  _Float16* __restrict__ out, int n) {
    int i = blockIdx.x * blockDim.x + threadIdx.x;
    if (i < n) out[i] = (_Float16)in[i];
}

// ---------------- branch 0: conv-bn-relu6 on full x, then global mean ----------------
__global__ __launch_bounds__(256) void pool0_kernel(
    const float* __restrict__ x, const float* __restrict__ pw,
    const float* __restrict__ gamma, const float* __restrict__ beta,
    const float* __restrict__ mu, const float* __restrict__ var,
    float* __restrict__ g0)
{
    __shared__ float red[256];
    int bd = blockIdx.x;           // b*16 + d
    int b = bd >> 4, d = bd & 15;
    const float* w = pw + d * C_IN;                 // pool_w[0][d][:]
    float ga = gamma[d], be = beta[d], m = mu[d], rs = rsqrtf(var[d] + EPSC);
    float partial = 0.f;
    for (int pix = threadIdx.x; pix < HW; pix += 256) {
        float s = 0.f;
        #pragma unroll 8
        for (int c = 0; c < C_IN; ++c)
            s += x[(((size_t)b * C_IN + c) << 14) + pix] * w[c];
        float v = ga * (s - m) * rs + be;
        v = fminf(fmaxf(v, 0.f), 6.f);
        partial += v;
    }
    red[threadIdx.x] = partial;
    __syncthreads();
    for (int st = 128; st > 0; st >>= 1) {
        if (threadIdx.x < st) red[threadIdx.x] += red[threadIdx.x + st];
        __syncthreads();
    }
    if (threadIdx.x == 0) g0[bd] = red[0] / (float)HW;
}

// ---------------- adaptive avg pool to s x s for s in {33,65,97} ----------------
__global__ __launch_bounds__(256) void adaptive_pool_kernel(
    const float* __restrict__ x, float* __restrict__ p)
{
    long idx = (long)blockIdx.x * 256 + threadIdx.x;
    if (idx >= (long)B_ * C_IN * APC) return;
    int e  = (int)(idx % APC);
    int bc = (int)(idx / APC);
    int s, area; long base;
    if (e < A1)           { s = S1; area = A1; base = 0; }
    else if (e < A1 + A2) { s = S2; area = A2; base = (long)B_ * C_IN * A1; e -= A1; }
    else                  { s = S3; area = A3; base = (long)B_ * C_IN * (A1 + A2); e -= A1 + A2; }
    int i = e / s, j = e % s;
    int h0 = (i * H_) / s, h1 = ((i + 1) * H_ + s - 1) / s;
    int w0 = (j * W_) / s, w1 = ((j + 1) * W_ + s - 1) / s;
    const float* xb = x + ((size_t)bc << 14);
    float sum = 0.f;
    for (int h = h0; h < h1; ++h)
        for (int w = w0; w < w1; ++w)
            sum += xb[(h << 7) + w];
    p[base + (size_t)bc * area + i * s + j] = sum / (float)((h1 - h0) * (w1 - w0));
}

// ---------------- 1x1 conv + BN + relu6 on pooled maps ----------------
__global__ __launch_bounds__(256) void pool_conv_kernel(
    const float* __restrict__ p, const float* __restrict__ pool_w,
    const float* __restrict__ gamma, const float* __restrict__ beta,
    const float* __restrict__ mu, const float* __restrict__ var,
    float* __restrict__ q)
{
    long idx = (long)blockIdx.x * 256 + threadIdx.x;
    if (idx >= (long)B_ * DIM * APC) return;
    int e  = (int)(idx % APC);
    int bd = (int)(idx / APC);
    int b = bd >> 4, d = bd & 15;
    int area, sidx; long pbase, qbase;
    if (e < A1)           { area=A1; sidx=1; pbase=0;                     qbase=0; }
    else if (e < A1 + A2) { area=A2; sidx=2; pbase=(long)B_*C_IN*A1;      qbase=(long)B_*DIM*A1;      e -= A1; }
    else                  { area=A3; sidx=3; pbase=(long)B_*C_IN*(A1+A2); qbase=(long)B_*DIM*(A1+A2); e -= A1 + A2; }
    const float* w = pool_w + (sidx * DIM + d) * C_IN;
    float acc = 0.f;
    #pragma unroll 8
    for (int c = 0; c < C_IN; ++c)
        acc += p[pbase + ((size_t)(b * C_IN + c)) * area + e] * w[c];
    int gi = sidx * DIM + d;
    float v = gamma[gi] * (acc - mu[gi]) * rsqrtf(var[gi] + EPSC) + beta[gi];
    v = fminf(fmaxf(v, 0.f), 6.f);
    q[qbase + (size_t)(b * DIM + d) * area + e] = v;
}

// ---------------- assemble seq (B,L,128) as f16: [x | g0 | up33 | up65 | up97] ----------------
__global__ __launch_bounds__(128) void assemble_kernel(
    const float* __restrict__ x, const float* __restrict__ g0,
    const float* __restrict__ q, _Float16* __restrict__ seq16)
{
    int bl = blockIdx.x;                    // token index b*HW + l
    int c  = threadIdx.x;                   // channel 0..127
    int b = bl >> 14, l = bl & (HW - 1);
    int h = l >> 7, w = l & 127;
    float v;
    if (c < 64) {
        v = x[(((size_t)b * C_IN + c) << 14) + l];
    } else if (c < 80) {
        v = g0[b * DIM + (c - 64)];
    } else {
        int grp = (c - 80) >> 4;            // 0,1,2 -> scale 33/65/97
        int d   = (c - 80) & 15;
        int s, area; long qbase;
        if (grp == 0)      { s=S1; area=A1; qbase=0; }
        else if (grp == 1) { s=S2; area=A2; qbase=(long)B_*DIM*A1; }
        else               { s=S3; area=A3; qbase=(long)B_*DIM*(A1+A2); }
        const float* qb = q + qbase + (size_t)(b * DIM + d) * area;
        float fs = (float)s / (float)H_;
        float fh = fminf(fmaxf((h + 0.5f) * fs - 0.5f, 0.f), (float)(s - 1));
        float fw = fminf(fmaxf((w + 0.5f) * fs - 0.5f, 0.f), (float)(s - 1));
        int h0 = (int)fh, w0 = (int)fw;
        int h1 = min(h0 + 1, s - 1), w1 = min(w0 + 1, s - 1);
        float ah = fh - (float)h0, aw = fw - (float)w0;
        float v00 = qb[h0 * s + w0], v01 = qb[h0 * s + w1];
        float v10 = qb[h1 * s + w0], v11 = qb[h1 * s + w1];
        v = (v00 * (1.f - aw) + v01 * aw) * (1.f - ah)
          + (v10 * (1.f - aw) + v11 * aw) * ah;
    }
    seq16[((size_t)bl << 7) + c] = (_Float16)v;
}

// ---------------- generic WMMA GEMM: C(MxNreal) = A(MxK,f16) * W(NrealxK,f16)^T ----
// MODE 0: row-major store (ld=Nreal).  MODE 1: out_proj store fused with
// (B,L,D_MODEL) -> (B,D_MODEL,H,W) transpose into d_out.
// GUARD: compile-time N-edge guard (only the x_proj GEMM needs it).
template<int MODE, bool GUARD>
__global__ __launch_bounds__(128) void wmma_gemm_kernel(
    const _Float16* __restrict__ A, const _Float16* __restrict__ W,
    float* __restrict__ C, int M, int tilesN, int Nreal, int K)
{
    int wave = threadIdx.x >> 5;
    int lane = threadIdx.x & 31;
    long tile = (long)blockIdx.x * 4 + wave;
    long totalTiles = (long)(M >> 4) * tilesN;
    if (tile >= totalTiles) return;                 // wave-uniform exit
    int  tn = (int)(tile % tilesN);
    long tm = tile / tilesN;

    // A 16-bit 16x32 layout (ISA 7.12.2): lanes 0-15 M=lane (K-half 0),
    // lanes 16-31 M=lane-16 (K-half 1); VGPR0-3: K in [0,16), VGPR4-7: K in [16,32)
    int mr = lane & 15;
    int kh = (lane >> 4) << 3;
    const _Float16* Ap = A + (size_t)(tm * 16 + mr) * K;
    int brow = tn * 16 + mr;
    bool bvalid = !GUARD || (brow < Nreal);
    const _Float16* Wp = W + (size_t)(GUARD ? (bvalid ? brow : 0) : brow) * K;

    v8f acc = {};
    for (int k0 = 0; k0 < K; k0 += 32) {
        __builtin_prefetch(Ap + k0 + 64, 0, 1);     // next A block -> global_prefetch
        v16h a, b;
        #pragma unroll
        for (int v = 0; v < 4; ++v) {
            a[2*v]     = Ap[k0 + kh + 2*v];
            a[2*v + 1] = Ap[k0 + kh + 2*v + 1];
            a[8 + 2*v] = Ap[k0 + 16 + kh + 2*v];
            a[9 + 2*v] = Ap[k0 + 16 + kh + 2*v + 1];
        }
        if (!GUARD || bvalid) {
            #pragma unroll
            for (int v = 0; v < 4; ++v) {
                b[2*v]     = Wp[k0 + kh + 2*v];
                b[2*v + 1] = Wp[k0 + kh + 2*v + 1];
                b[8 + 2*v] = Wp[k0 + 16 + kh + 2*v];
                b[9 + 2*v] = Wp[k0 + 16 + kh + 2*v + 1];
            }
        } else {
            #pragma unroll
            for (int i = 0; i < 16; ++i) b[i] = (_Float16)0.f;
        }
        acc = __builtin_amdgcn_wmma_f32_16x16x32_f16(
                  false, a, false, b, (short)0, acc, false, false);
    }

    // C/D layout: VGPR r -> M = r (+8 for lanes 16-31), N = lane&15
    int colOff = lane & 15;
    int rowOff = (lane >> 4) << 3;
    #pragma unroll
    for (int r = 0; r < 8; ++r) {
        long row = tm * 16 + rowOff + r;
        int  col = tn * 16 + colOff;
        if (!GUARD || col < Nreal) {
            if (MODE == 0) {
                C[(size_t)row * Nreal + col] = acc[r];
            } else {
                long bb = row >> 14;             // batch
                long l  = row & (HW - 1);        // pixel
                C[((bb * D_MODEL + col) << 14) + l] = acc[r];
            }
        }
    }
}

// ---------------- causal depthwise conv1d (k=4) + SiLU on xi ----------------
__global__ __launch_bounds__(256) void conv1d_silu_kernel(
    const float* __restrict__ xz, const float* __restrict__ cw,
    const float* __restrict__ cb, float* __restrict__ xt32,
    _Float16* __restrict__ xt16)
{
    long idx = (long)blockIdx.x * 256 + threadIdx.x;   // < M_TOK*256
    int  d  = (int)(idx & 255);
    long bl = idx >> 8;
    int  l  = (int)(bl & (HW - 1));
    long b  = bl >> 14;
    const float* wd = cw + d * 4;
    float acc = cb[d];
    #pragma unroll
    for (int k = 0; k < 4; ++k) {
        int ll = l - 3 + k;
        if (ll >= 0)
            acc += wd[k] * xz[(((b << 14) + ll) << 9) + d];   // xi = xz[:, :256]
    }
    float sv = acc / (1.f + __expf(-acc));
    xt32[idx] = sv;
    xt16[idx] = (_Float16)sv;
}

// ---------------- delta = softplus(dt @ dt_proj_w^T + b), K=8 scalar ----------------
__global__ __launch_bounds__(256) void delta_kernel(
    const float* __restrict__ dbl, const float* __restrict__ dtw,
    const float* __restrict__ dtb, float* __restrict__ delta)
{
    long idx = (long)blockIdx.x * 256 + threadIdx.x;   // < M_TOK*256
    int  d = (int)(idx & 255);
    long m = idx >> 8;
    const float* w = dtw + d * DT_RANK;
    const float* r = dbl + m * 40;                     // cols 0..7 are dt
    float acc = dtb[d];
    #pragma unroll
    for (int k = 0; k < DT_RANK; ++k) acc += w[k] * r[k];
    delta[idx] = (acc > 20.f) ? acc : log1pf(__expf(acc));
}

// ================= chunked selective scan =================
// Recurrence h_l = dA_l*h_{l-1} + (dt_l*x_l)*B_l per (b,d,n).
// Pass 1: per chunk, local scan from h=0 -> (prod dA, local end state).
// Pass 2: tiny cross-chunk combine -> incoming state per chunk.
// Pass 3: re-run each chunk from its true incoming state, emit gated output.

// Pass 1: grid = B_*NCHUNK blocks, 256 threads (d)
__global__ __launch_bounds__(256) void scan_pass1_kernel(
    const float* __restrict__ delta, const float* __restrict__ xt,
    const float* __restrict__ dbl,   const float* __restrict__ A_log,
    float* __restrict__ Aprod, float* __restrict__ Ssum)
{
    int bc = blockIdx.x;                 // b*NCHUNK + c
    int b = bc >> 7, c = bc & (NCHUNK - 1);
    int d = threadIdx.x;
    float An[D_STATE], h[D_STATE], P[D_STATE];
    #pragma unroll
    for (int n = 0; n < D_STATE; ++n) {
        An[n] = -__expf(A_log[d * D_STATE + n]);
        h[n] = 0.f;
        P[n] = 1.f;
    }
    size_t base = ((size_t)b << 14) + (size_t)c * CHUNK;
    for (int i = 0; i < CHUNK; ++i) {
        size_t t = base + i;
        float dt = delta[(t << 8) + d];
        float xv = xt[(t << 8) + d];
        float dx = dt * xv;
        const float* Bt = dbl + t * 40 + DT_RANK;
        #pragma unroll
        for (int n = 0; n < D_STATE; ++n) {
            float dA = __expf(dt * An[n]);
            h[n] = dA * h[n] + dx * Bt[n];
            P[n] *= dA;
        }
    }
    size_t o = ((size_t)bc * 256 + d) * D_STATE;
    #pragma unroll
    for (int n = 0; n < D_STATE; ++n) {
        Aprod[o + n] = P[n];
        Ssum[o + n]  = h[n];
    }
}

// Pass 2: 1024 threads total (b,d); 128-step serial combine (tiny)
__global__ __launch_bounds__(256) void scan_pass2_kernel(
    const float* __restrict__ Aprod, const float* __restrict__ Ssum,
    float* __restrict__ Hin)
{
    int idx = blockIdx.x * 256 + threadIdx.x;   // < B_*256
    int b = idx >> 8, d = idx & 255;
    float H[D_STATE];
    #pragma unroll
    for (int n = 0; n < D_STATE; ++n) H[n] = 0.f;
    for (int c = 0; c < NCHUNK; ++c) {
        size_t o = ((size_t)(b * NCHUNK + c) * 256 + d) * D_STATE;
        #pragma unroll
        for (int n = 0; n < D_STATE; ++n) {
            Hin[o + n] = H[n];
            H[n] = Aprod[o + n] * H[n] + Ssum[o + n];
        }
    }
}

// Pass 3: grid = B_*NCHUNK blocks, 256 threads; re-scan from Hin, gate, emit f16
__global__ __launch_bounds__(256) void scan_pass3_kernel(
    const float* __restrict__ delta, const float* __restrict__ xt,
    const float* __restrict__ dbl,   const float* __restrict__ xz,
    const float* __restrict__ A_log, const float* __restrict__ Dp,
    const float* __restrict__ Hin,   _Float16* __restrict__ yg)
{
    int bc = blockIdx.x;
    int b = bc >> 7, c = bc & (NCHUNK - 1);
    int d = threadIdx.x;
    float An[D_STATE], h[D_STATE];
    size_t o = ((size_t)bc * 256 + d) * D_STATE;
    #pragma unroll
    for (int n = 0; n < D_STATE; ++n) {
        An[n] = -__expf(A_log[d * D_STATE + n]);
        h[n] = Hin[o + n];
    }
    float Dd = Dp[d];
    size_t base = ((size_t)b << 14) + (size_t)c * CHUNK;
    for (int i = 0; i < CHUNK; ++i) {
        size_t t = base + i;
        float dt = delta[(t << 8) + d];
        float xv = xt[(t << 8) + d];
        float dx = dt * xv;
        const float* Bt = dbl + t * 40 + DT_RANK;
        const float* Ct = Bt + D_STATE;
        float y = 0.f;
        #pragma unroll
        for (int n = 0; n < D_STATE; ++n) {
            float dA = __expf(dt * An[n]);
            h[n] = dA * h[n] + dx * Bt[n];
            y += h[n] * Ct[n];
        }
        float zv = xz[(t << 9) + D_INNER + d];
        float out = (y + xv * Dd) * (zv / (1.f + __expf(-zv)));
        yg[(t << 8) + d] = (_Float16)out;
    }
}

// ---------------- launcher ----------------
extern "C" void kernel_launch(void* const* d_in, const int* in_sizes, int n_in,
                              void* d_out, int out_size, void* d_ws, size_t ws_size,
                              hipStream_t stream)
{
    const float* x          = (const float*)d_in[0];
    const float* pool_w     = (const float*)d_in[1];
    const float* pool_gamma = (const float*)d_in[2];
    const float* pool_beta  = (const float*)d_in[3];
    const float* pool_mu    = (const float*)d_in[4];
    const float* pool_var   = (const float*)d_in[5];
    const float* in_proj_w  = (const float*)d_in[6];
    const float* conv1d_w   = (const float*)d_in[7];
    const float* conv1d_b   = (const float*)d_in[8];
    const float* x_proj_w   = (const float*)d_in[9];
    const float* dt_proj_w  = (const float*)d_in[10];
    const float* dt_proj_b  = (const float*)d_in[11];
    const float* A_log      = (const float*)d_in[12];
    const float* Dw         = (const float*)d_in[13];
    const float* out_proj_w = (const float*)d_in[14];

    char* ws = (char*)d_ws;
    size_t off = 0;
    auto alloc = [&](size_t bytes) -> void* {
        void* p = ws + off;
        off += (bytes + 255) & ~(size_t)255;
        return p;
    };
    float*     p_pool  = (float*)    alloc((size_t)B_ * C_IN * APC * 4);   // ~15 MB
    float*     q_pool  = (float*)    alloc((size_t)B_ * DIM  * APC * 4);   // ~3.8 MB
    float*     g0      = (float*)    alloc(B_ * DIM * 4);
    _Float16*  seq16   = (_Float16*) alloc((size_t)M_TOK * D_MODEL * 2);   // 16 MB
    _Float16*  w_in16  = (_Float16*) alloc((size_t)512 * 128 * 2);
    _Float16*  w_xp16  = (_Float16*) alloc((size_t)40 * 256 * 2);
    _Float16*  w_out16 = (_Float16*) alloc((size_t)128 * 256 * 2);
    float*     xz      = (float*)    alloc((size_t)M_TOK * 512 * 4);       // 128 MB
    float*     xt32    = (float*)    alloc((size_t)M_TOK * 256 * 4);       // 64 MB
    _Float16*  xt16    = (_Float16*) alloc((size_t)M_TOK * 256 * 2);       // 32 MB
    float*     dbl     = (float*)    alloc((size_t)M_TOK * 40 * 4);        // 10.5 MB
    float*     delta   = (float*)    alloc((size_t)M_TOK * 256 * 4);       // 64 MB
    _Float16*  yg16    = (_Float16*) alloc((size_t)M_TOK * 256 * 2);       // 32 MB
    size_t     scanN   = (size_t)B_ * NCHUNK * 256 * D_STATE;              // 2M
    float*     Aprod   = (float*)    alloc(scanN * 4);                     // 8 MB
    float*     Ssum    = (float*)    alloc(scanN * 4);                     // 8 MB
    float*     Hin     = (float*)    alloc(scanN * 4);                     // 8 MB

    // weight conversions f32 -> f16
    f32_to_f16_kernel<<<(512*128 + 255)/256, 256, 0, stream>>>(in_proj_w,  w_in16,  512*128);
    f32_to_f16_kernel<<<(40*256  + 255)/256, 256, 0, stream>>>(x_proj_w,   w_xp16,  40*256);
    f32_to_f16_kernel<<<(128*256 + 255)/256, 256, 0, stream>>>(out_proj_w, w_out16, 128*256);

    // PSP frontend
    pool0_kernel<<<B_ * DIM, 256, 0, stream>>>(x, pool_w, pool_gamma, pool_beta,
                                               pool_mu, pool_var, g0);
    long npool = (long)B_ * C_IN * APC;
    adaptive_pool_kernel<<<(int)((npool + 255) / 256), 256, 0, stream>>>(x, p_pool);
    long nq = (long)B_ * DIM * APC;
    pool_conv_kernel<<<(int)((nq + 255) / 256), 256, 0, stream>>>(
        p_pool, pool_w, pool_gamma, pool_beta, pool_mu, pool_var, q_pool);
    assemble_kernel<<<M_TOK, 128, 0, stream>>>(x, g0, q_pool, seq16);

    // in_proj: (65536 x 128) x (512 x 128)^T -> xz   (no N guard)
    {
        long tiles = (long)(M_TOK / 16) * (512 / 16);
        wmma_gemm_kernel<0, false><<<(int)((tiles + 3) / 4), 128, 0, stream>>>(
            seq16, w_in16, xz, M_TOK, 512 / 16, 512, 128);
    }
    // depthwise conv + SiLU
    conv1d_silu_kernel<<<(int)((size_t)M_TOK * 256 / 256), 256, 0, stream>>>(
        xz, conv1d_w, conv1d_b, xt32, xt16);
    // x_proj: (65536 x 256) x (40 x 256)^T -> dbl (N=40 padded to 3 tiles, guarded)
    {
        long tiles = (long)(M_TOK / 16) * 3;
        wmma_gemm_kernel<0, true><<<(int)((tiles + 3) / 4), 128, 0, stream>>>(
            xt16, w_xp16, dbl, M_TOK, 3, 40, 256);
    }
    // delta
    delta_kernel<<<(int)((size_t)M_TOK * 256 / 256), 256, 0, stream>>>(
        dbl, dt_proj_w, dt_proj_b, delta);
    // chunked selective scan + D skip + SiLU gate
    scan_pass1_kernel<<<B_ * NCHUNK, 256, 0, stream>>>(
        delta, xt32, dbl, A_log, Aprod, Ssum);
    scan_pass2_kernel<<<B_, 256, 0, stream>>>(Aprod, Ssum, Hin);
    scan_pass3_kernel<<<B_ * NCHUNK, 256, 0, stream>>>(
        delta, xt32, dbl, xz, A_log, Dw, Hin, yg16);
    // out_proj with fused transpose into d_out (B, 128, H, W)   (no N guard)
    {
        long tiles = (long)(M_TOK / 16) * (128 / 16);
        wmma_gemm_kernel<1, false><<<(int)((tiles + 3) / 4), 128, 0, stream>>>(
            yg16, w_out16, (float*)d_out, M_TOK, 128 / 16, 128, 256);
    }
    (void)in_sizes; (void)n_in; (void)out_size; (void)ws_size;
}
